// GAT_2284922601516
// MI455X (gfx1250) — compile-verified, matrix operand order checked by reading, not run on hardware
//
#include <hip/hip_runtime.h>
#include <math.h>

// ---------------- problem constants (match reference) ----------------
#define NN      50000     // nodes
#define IN_CH   256
#define HID     64
#define HEADS   4
#define OUT_CH  40
#define NEG_SLOPE 0.2f
#define BN_EPS  1e-5f

typedef __attribute__((ext_vector_type(16))) _Float16 v16h;
typedef __attribute__((ext_vector_type(8)))  float    v8f;

// ------- float <-> order-preserving uint (for atomicMax on floats) -------
__device__ __forceinline__ unsigned f2ord(float f) {
    unsigned u = __float_as_uint(f);
    return (u & 0x80000000u) ? ~u : (u | 0x80000000u);
}
__device__ __forceinline__ float ord2f(unsigned u) {
    unsigned b = (u & 0x80000000u) ? (u ^ 0x80000000u) : ~u;
    return __uint_as_float(b);
}

// =====================================================================
// pack_B: gather W[K,NCOLS] (f32, row-major) into per-lane B-fragment
// layout (f16): one contiguous v16h per (tile, k-step, lane).
//   Wp[((nt*KS + ks)*32 + lane)*16 + j]
//     j<8  -> W[(ks*32 +      hl*8 + j)*NCOLS + col]
//     j>=8 -> W[(ks*32 + 16 + hl*8 + (j-8))*NCOLS + col]
//   col = clamp(nt*16 + (lane&15));  hl = lane>>4
// =====================================================================
template<int K, int NCOLS>
__global__ void pack_B(const float* __restrict__ W, _Float16* __restrict__ Wp)
{
    constexpr int NT = (NCOLS + 15) / 16;
    constexpr int KS = K / 32;
    int tid = blockIdx.x * blockDim.x + threadIdx.x;
    if (tid >= NT * KS * 32) return;
    const int lane = tid & 31;
    const int ks   = (tid >> 5) % KS;
    const int nt   = tid / (32 * KS);
    const int mr = lane & 15, hl = lane >> 4;
    int col = nt * 16 + mr;
    if (col >= NCOLS) col = NCOLS - 1;            // clamp; lane never stored
    _Float16* out = Wp + (size_t)tid * 16;
    #pragma unroll
    for (int j = 0; j < 8; ++j) {
        out[j]     = (_Float16)W[(ks * 32 +      hl * 8 + j) * NCOLS + col];
        out[j + 8] = (_Float16)W[(ks * 32 + 16 + hl * 8 + j) * NCOLS + col];
    }
}

// =====================================================================
// GEMM: C[M,NCOLS] = A[M,K] * B  (A f32, B pre-packed f16, f32 WMMA acc)
// One wave computes FOUR 16-wide N-tiles (4 accumulators) so each A
// fragment load feeds 4 WMMAs (4x less A traffic from L2).
//  - NCOLS multiple of 64: blockDim=128 (4 waves cover all columns), grid.y=1
//  - otherwise: blockDim=32, single wave walks the NT (<4) tiles,
//    guards fold at compile time -> no divergence around WMMA.
// =====================================================================
template<int K, int NCOLS>
__global__ void gemm_wmma_f16(const float* __restrict__ A,
                              const _Float16* __restrict__ Bp,
                              float* __restrict__ C)
{
    constexpr int KS = K / 32;
    constexpr int NT = (NCOLS + 15) / 16;
    constexpr bool FULL = (NT % 4 == 0);          // every wave has 4 valid tiles
    const int lane = threadIdx.x & 31;
    const int wave = threadIdx.x >> 5;
    const int hl   = lane >> 4;
    const int mr   = lane & 15;
    const int nt0  = FULL ? wave * 4 : 0;         // non-FULL path runs 1 wave

    const int row = blockIdx.x * 16 + mr;
    const float* arow = A + (size_t)row * K + hl * 8;

    v8f acc[4] = {};
    #pragma unroll
    for (int ks = 0; ks < KS; ++ks) {
        const float* ap = arow + ks * 32;
        v16h a;
        #pragma unroll
        for (int j = 0; j < 8; ++j) {
            a[j]     = (_Float16)ap[j];
            a[j + 8] = (_Float16)ap[16 + j];
        }
        #pragma unroll
        for (int t = 0; t < 4; ++t) {
            if (FULL || t < NT) {                 // compile-time guard
                const int nt = nt0 + t;
                v16h b = *(const v16h*)(Bp + ((size_t)(nt * KS + ks) * 32 + lane) * 16);
                acc[t] = __builtin_amdgcn_wmma_f32_16x16x32_f16(
                    false, a, false, b, (short)0, acc[t], false, false);
            }
        }
    }
    #pragma unroll
    for (int t = 0; t < 4; ++t) {
        if (FULL || t < NT) {
            const int col = (nt0 + t) * 16 + mr;
            #pragma unroll
            for (int r = 0; r < 8; ++r) {
                int orow = blockIdx.x * 16 + hl * 8 + r;
                if (col < NCOLS) C[(size_t)orow * NCOLS + col] = acc[t][r];
            }
        }
    }
}

// =====================================================================
// Per-node prep: attention scalars, zero accumulator, init max/denom.
// One block (256 threads) per node.
// =====================================================================
template<int CH, int Hh, int Cper>
__global__ void node_prep(const float* __restrict__ H,
                          const float* __restrict__ att_s,
                          const float* __restrict__ att_d,
                          float* __restrict__ acc,
                          float* __restrict__ asrc, float* __restrict__ adst,
                          unsigned* __restrict__ mmax, float* __restrict__ den)
{
    const int n = blockIdx.x;
    const int t = threadIdx.x;
    if (t < CH) acc[(size_t)n * CH + t] = 0.0f;
    if (t < Hh) {
        const float* hrow = H + (size_t)n * CH + t * Cper;
        float s = 0.0f, d = 0.0f;
        #pragma unroll 4
        for (int c = 0; c < Cper; ++c) {
            float hv = hrow[c];
            s += hv * att_s[t * Cper + c];
            d += hv * att_d[t * Cper + c];
        }
        asrc[n * Hh + t] = s;
        adst[n * Hh + t] = d;
        mmax[n * Hh + t] = 0u;      // ordered encoding below any real value
        den [n * Hh + t] = 0.0f;
    }
}

// =====================================================================
// Edge pass 1: segment max of leaky_relu(a_s[src] + a_d[dst])
// =====================================================================
template<int Hh>
__global__ void edge_max(const int* __restrict__ ei, int E,
                         const float* __restrict__ asrc,
                         const float* __restrict__ adst,
                         unsigned* __restrict__ mmax)
{
    int tid = blockIdx.x * blockDim.x + threadIdx.x;
    if (tid >= E * Hh) return;
    int e = tid / Hh, h = tid - e * Hh;
    int src = ei[e], dst = ei[E + e];
    float a  = asrc[src * Hh + h] + adst[dst * Hh + h];
    float ev = (a > 0.0f) ? a : NEG_SLOPE * a;
    atomicMax(&mmax[dst * Hh + h], f2ord(ev));
}

// =====================================================================
// Edge pass 2: p = exp(e - m[dst]); denom += p; acc[dst,c] += h[src,c]*p
// one wave per edge; float4 (b128) gathers from L2-resident h.
// =====================================================================
template<int Hh, int CH>
__global__ void edge_acc(const int* __restrict__ ei, int E,
                         const float* __restrict__ Hbuf,
                         const float* __restrict__ asrc,
                         const float* __restrict__ adst,
                         const unsigned* __restrict__ mmax,
                         float* __restrict__ den,
                         float* __restrict__ acc)
{
    int wid  = (int)((blockIdx.x * (size_t)blockDim.x + threadIdx.x) >> 5);
    int lane = threadIdx.x & 31;
    if (wid >= E) return;
    int src = ei[wid], dst = ei[E + wid];

    const float* hs = Hbuf + (size_t)src * CH;
    float*       ad = acc  + (size_t)dst * CH;
    __builtin_prefetch(hs, 0, 3);                 // global_prefetch (L2-hot row)

    float p[Hh];
    #pragma unroll
    for (int h = 0; h < Hh; ++h) {
        float a  = asrc[src * Hh + h] + adst[dst * Hh + h];
        float ev = (a > 0.0f) ? a : NEG_SLOPE * a;
        float m  = ord2f(mmax[dst * Hh + h]);
        p[h] = __expf(ev - m);
    }
    if (lane < Hh) {
        float pv = p[0];
        #pragma unroll
        for (int h = 1; h < Hh; ++h) if (lane == h) pv = p[h];
        atomicAdd(&den[dst * Hh + lane], pv);
    }
    #pragma unroll
    for (int c0 = 0; c0 < CH; c0 += 128) {
        int c = c0 + lane * 4;
        if (c < CH) {
            float4 v = *(const float4*)(hs + c);
            float pv = p[0];
            if (Hh > 1) {                          // Cper == 64 when Hh == 4
                int h = c >> 6;
                pv = (h == 0) ? p[0] : ((h == 1) ? p[1] : ((h == 2) ? p[2] : p[3]));
            }
            atomicAdd(&ad[c + 0], v.x * pv);
            atomicAdd(&ad[c + 1], v.y * pv);
            atomicAdd(&ad[c + 2], v.z * pv);
            atomicAdd(&ad[c + 3], v.w * pv);
        }
    }
}

// =====================================================================
// Per-element epilogue: normalize by denom, +bias, optional BN+ELU
// =====================================================================
template<int CH, int Hh, int Cper, bool DO_BN>
__global__ void post_epilogue(const float* __restrict__ acc,
                              const float* __restrict__ den,
                              const float* __restrict__ bias,
                              const float* __restrict__ g,
                              const float* __restrict__ bb,
                              const float* __restrict__ bm,
                              const float* __restrict__ bv,
                              float* __restrict__ outp)
{
    size_t idx = (size_t)blockIdx.x * blockDim.x + threadIdx.x;
    if (idx >= (size_t)NN * CH) return;
    int n = (int)(idx / CH);
    int c = (int)(idx - (size_t)n * CH);
    float v = acc[idx] / (den[n * Hh + c / Cper] + 1e-16f) + bias[c];
    if (DO_BN) {
        v = g[c] * (v - bm[c]) * rsqrtf(bv[c] + BN_EPS) + bb[c];
        v = (v > 0.0f) ? v : (__expf(v) - 1.0f);   // ELU
    }
    outp[idx] = v;
}

// =====================================================================
// Host launch
// =====================================================================
extern "C" void kernel_launch(void* const* d_in, const int* in_sizes, int n_in,
                              void* d_out, int out_size, void* d_ws, size_t ws_size,
                              hipStream_t stream)
{
    const float* x    = (const float*)d_in[0];
    const int*   ei   = (const int*)  d_in[1];
    const int    E    = in_sizes[1] / 2;          // 850000
    const float* W0   = (const float*)d_in[2];
    const float* as0  = (const float*)d_in[3];
    const float* ad0  = (const float*)d_in[4];
    const float* b0   = (const float*)d_in[5];
    const float* bn0g = (const float*)d_in[6];
    const float* bn0b = (const float*)d_in[7];
    const float* bn0m = (const float*)d_in[8];
    const float* bn0v = (const float*)d_in[9];
    const float* W1   = (const float*)d_in[10];
    const float* as1  = (const float*)d_in[11];
    const float* ad1  = (const float*)d_in[12];
    const float* b1   = (const float*)d_in[13];
    const float* bn1g = (const float*)d_in[14];
    const float* bn1b = (const float*)d_in[15];
    const float* bn1m = (const float*)d_in[16];
    const float* bn1v = (const float*)d_in[17];
    const float* W2   = (const float*)d_in[18];
    const float* as2  = (const float*)d_in[19];
    const float* ad2  = (const float*)d_in[20];
    const float* b2   = (const float*)d_in[21];
    float* outp = (float*)d_out;

    // ---- workspace layout ----
    float*    ws    = (float*)d_ws;
    float*    Hbuf  = ws;                                     // [NN,256]
    float*    ACC   = Hbuf + (size_t)NN * IN_CH;              // [NN,256]
    float*    ASRC  = ACC  + (size_t)NN * IN_CH;              // [NN,4]
    float*    ADST  = ASRC + (size_t)NN * HEADS;              // [NN,4]
    unsigned* MMAX  = (unsigned*)(ADST + (size_t)NN * HEADS); // [NN,4]
    float*    DEN   = (float*)MMAX + (size_t)NN * HEADS;      // [NN,4]
    _Float16* Wp0   = (_Float16*)(DEN + (size_t)NN * HEADS);  // 16*8*32*16
    _Float16* Wp1   = Wp0 + 16 * 8 * 32 * 16;
    _Float16* Wp2   = Wp1 + 16 * 8 * 32 * 16;                 // 3*8*32*16

    const int  mTiles      = NN / 16;                         // 3125
    const int  emaxBlocks4 = (E * HEADS + 255) / 256;
    const int  emaxBlocks1 = (E + 255) / 256;
    const int  eaccBlocks  = (E + 7) / 8;                     // 8 waves / block
    const int  postB256    = (int)(((size_t)NN * IN_CH + 255) / 256);
    const int  postB40     = (int)(((size_t)NN * OUT_CH + 255) / 256);

    // ---- pack weights into WMMA B-fragment layout (every call) ----
    pack_B<IN_CH, IN_CH> <<<(16 * 8 * 32 + 255) / 256, 256, 0, stream>>>(W0, Wp0);
    pack_B<IN_CH, IN_CH> <<<(16 * 8 * 32 + 255) / 256, 256, 0, stream>>>(W1, Wp1);
    pack_B<IN_CH, OUT_CH><<<(3 * 8 * 32 + 255) / 256, 256, 0, stream>>>(W2, Wp2);

    // ---------------- Layer 0 ----------------
    gemm_wmma_f16<IN_CH, IN_CH><<<mTiles, 128, 0, stream>>>(x, Wp0, Hbuf);
    node_prep<IN_CH, HEADS, HID><<<NN, 256, 0, stream>>>(Hbuf, as0, ad0, ACC,
                                                         ASRC, ADST, MMAX, DEN);
    edge_max<HEADS><<<emaxBlocks4, 256, 0, stream>>>(ei, E, ASRC, ADST, MMAX);
    edge_acc<HEADS, IN_CH><<<eaccBlocks, 256, 0, stream>>>(ei, E, Hbuf, ASRC, ADST,
                                                           MMAX, DEN, ACC);
    post_epilogue<IN_CH, HEADS, HID, true><<<postB256, 256, 0, stream>>>(
        ACC, DEN, b0, bn0g, bn0b, bn0m, bn0v, ACC);

    // ---------------- Layer 1 ----------------
    gemm_wmma_f16<IN_CH, IN_CH><<<mTiles, 128, 0, stream>>>(ACC, Wp1, Hbuf);
    node_prep<IN_CH, HEADS, HID><<<NN, 256, 0, stream>>>(Hbuf, as1, ad1, ACC,
                                                         ASRC, ADST, MMAX, DEN);
    edge_max<HEADS><<<emaxBlocks4, 256, 0, stream>>>(ei, E, ASRC, ADST, MMAX);
    edge_acc<HEADS, IN_CH><<<eaccBlocks, 256, 0, stream>>>(ei, E, Hbuf, ASRC, ADST,
                                                           MMAX, DEN, ACC);
    post_epilogue<IN_CH, HEADS, HID, true><<<postB256, 256, 0, stream>>>(
        ACC, DEN, b1, bn1g, bn1b, bn1m, bn1v, ACC);

    // -------- Layer 2 (1 head, 40 channels, no BN/ELU) --------
    gemm_wmma_f16<IN_CH, OUT_CH><<<mTiles, 32, 0, stream>>>(ACC, Wp2, Hbuf);
    node_prep<OUT_CH, 1, OUT_CH><<<NN, 256, 0, stream>>>(Hbuf, as2, ad2, ACC,
                                                         ASRC, ADST, MMAX, DEN);
    edge_max<1><<<emaxBlocks1, 256, 0, stream>>>(ei, E, ASRC, ADST, MMAX);
    edge_acc<1, OUT_CH><<<eaccBlocks, 256, 0, stream>>>(ei, E, Hbuf, ASRC, ADST,
                                                        MMAX, DEN, ACC);
    post_epilogue<OUT_CH, 1, OUT_CH, false><<<postB40, 256, 0, stream>>>(
        ACC, DEN, b2, nullptr, nullptr, nullptr, nullptr, outp);
}
